// Attention_6734508720309
// MI455X (gfx1250) — matrix operand order repaired and checked by reading, not compiled, not run on hardware
//
#include <hip/hip_runtime.h>
#include <math.h>

#define B_ 2
#define S_ 2048
#define HID_ 4096
#define NQ_ 32
#define NK_ 8
#define HD_ 128
#define WINDOW_ 1024

typedef __bf16 bf16;
typedef __attribute__((ext_vector_type(16))) bf16 v16bf;
typedef __attribute__((ext_vector_type(8)))  bf16 v8bf;
typedef __attribute__((ext_vector_type(8)))  float v8f;

union BF16x16 { v16bf v; v8bf h[2]; };

__device__ __forceinline__ bf16 f2bf(float f) {
  unsigned u = __builtin_bit_cast(unsigned, f);
  unsigned r = u + 0x7FFFu + ((u >> 16) & 1u);
  unsigned short hs = (unsigned short)(r >> 16);
  return __builtin_bit_cast(bf16, hs);
}

__device__ __forceinline__ v8f v8f_zero() {
  v8f z;
#pragma unroll
  for (int e = 0; e < 8; ++e) z[e] = 0.0f;
  return z;
}

// ---------------- fp32 -> bf16 convert ----------------
__global__ void f32_to_bf16_kernel(const float* __restrict__ src, bf16* __restrict__ dst, int n) {
  int i = blockIdx.x * blockDim.x + threadIdx.x;
  if (i < n) dst[i] = f2bf(src[i]);
}

// ---------------- GEMM: C[M,N] = A[M,K] * W[N,K]^T  (bf16 in, f32 out) ----------------
// block = 256 threads = 8 waves in a 2(M) x 4(N) grid; wave tile 64x32 (4x2 WMMA tiles)
// block tile 128(M) x 128(N). 8 WMMAs per 6 fragment loads per k-step.
__global__ __launch_bounds__(256)
void gemm_bf16_wmma_kernel(const bf16* __restrict__ A, const bf16* __restrict__ W,
                           float* __restrict__ C, int M, int N, int K) {
  const int lane  = threadIdx.x & 31;
  const int wave  = threadIdx.x >> 5;
  const int m16   = lane & 15;
  const int half  = lane >> 4;
  const int mBase = blockIdx.y * 128 + (wave >> 2) * 64;
  const int nBase = blockIdx.x * 128 + (wave & 3) * 32;

  v8f acc[4][2];
#pragma unroll
  for (int tm = 0; tm < 4; ++tm)
#pragma unroll
    for (int tn = 0; tn < 2; ++tn) acc[tm][tn] = v8f_zero();

  for (int k = 0; k < K; k += 32) {
    BF16x16 a[4], b[2];
#pragma unroll
    for (int tm = 0; tm < 4; ++tm) {
      const bf16* ap = A + (size_t)(mBase + tm * 16 + m16) * K + k;
      a[tm].h[0] = *(const v8bf*)(ap + half * 8);
      a[tm].h[1] = *(const v8bf*)(ap + 16 + half * 8);
    }
#pragma unroll
    for (int tn = 0; tn < 2; ++tn) {
      const bf16* wp = W + (size_t)(nBase + tn * 16 + m16) * K + k;
      b[tn].h[0] = *(const v8bf*)(wp + half * 8);
      b[tn].h[1] = *(const v8bf*)(wp + 16 + half * 8);
    }
#pragma unroll
    for (int tm = 0; tm < 4; ++tm)
#pragma unroll
      for (int tn = 0; tn < 2; ++tn)
        acc[tm][tn] = __builtin_amdgcn_wmma_f32_16x16x32_bf16(
            false, a[tm].v, false, b[tn].v, (short)0, acc[tm][tn], false, false);
  }
#pragma unroll
  for (int tm = 0; tm < 4; ++tm)
#pragma unroll
    for (int tn = 0; tn < 2; ++tn)
#pragma unroll
      for (int e = 0; e < 8; ++e) {
        const int row = mBase + tm * 16 + half * 8 + e;
        const int col = nBase + tn * 16 + m16;
        C[(size_t)row * N + col] = acc[tm][tn][e];
      }
}

// ---------------- per-head RMSNorm + NeoX RoPE, fp32 -> bf16 ----------------
// one wave per (b, s, head); 4 dims per lane (HD=128)
__global__ __launch_bounds__(128)
void qknorm_rope_kernel(const float* __restrict__ src, bf16* __restrict__ dst,
                        const float* __restrict__ nw, int nheads) {
  const int lane = threadIdx.x & 31;
  const int idx  = blockIdx.x * 4 + (threadIdx.x >> 5);
  const int h  = idx % nheads;
  const int sp = (idx / nheads) % S_;
  const int b  = idx / (nheads * S_);
  const size_t base = (((size_t)b * S_ + sp) * nheads + h) * HD_;

  float e0 = src[base + lane];
  float e1 = src[base + lane + 32];
  float e2 = src[base + lane + 64];
  float e3 = src[base + lane + 96];

  float ss = e0 * e0 + e1 * e1 + e2 * e2 + e3 * e3;
#pragma unroll
  for (int off = 16; off > 0; off >>= 1) ss += __shfl_xor(ss, off, 32);
  const float r = rsqrtf(ss * (1.0f / HD_) + 1e-6f);

  float n0 = e0 * r * nw[lane];
  float n1 = e1 * r * nw[lane + 32];
  float n2 = e2 * r * nw[lane + 64];
  float n3 = e3 * r * nw[lane + 96];

  const float inv0 = powf(10000.0f, -(float)lane * (1.0f / 64.0f));
  const float inv1 = powf(10000.0f, -(float)(lane + 32) * (1.0f / 64.0f));
  const float a0 = (float)sp * inv0, a1 = (float)sp * inv1;
  const float c0 = cosf(a0), s0 = sinf(a0);
  const float c1 = cosf(a1), s1 = sinf(a1);

  dst[base + lane]      = f2bf(n0 * c0 - n2 * s0);
  dst[base + lane + 32] = f2bf(n1 * c1 - n3 * s1);
  dst[base + lane + 64] = f2bf(n2 * c0 + n0 * s0);
  dst[base + lane + 96] = f2bf(n3 * c1 + n1 * s1);
}

// ---------------- V: [B,S,NK,HD] fp32 -> [B,NK,HD,S] bf16 ----------------
__global__ void v_transpose_kernel(const float* __restrict__ src, bf16* __restrict__ dst) {
  const int tid = blockIdx.x * blockDim.x + threadIdx.x;
  const int total = B_ * NK_ * HD_ * S_;
  if (tid >= total) return;
  const int s  = tid % S_;
  const int d  = (tid / S_) % HD_;
  const int kh = (tid / (S_ * HD_)) % NK_;
  const int b  = tid / (S_ * HD_ * NK_);
  dst[tid] = f2bf(src[(((size_t)b * S_ + s) * NK_ + kh) * HD_ + d]);
}

// ---------------- flash attention: softcap + causal sliding window, GQA ----------------
// one wave handles one (b, head, 16-query tile); 4 waves per block
__global__ __launch_bounds__(128)
void flash_attn_kernel(const bf16* __restrict__ q, const bf16* __restrict__ kk,
                       const bf16* __restrict__ vt, float* __restrict__ o) {
  __shared__ __align__(16) bf16 ldsP[4][16 * 32];
  const int lane = threadIdx.x & 31;
  const int w    = threadIdx.x >> 5;
  const int m16  = lane & 15;
  const int half = lane >> 4;

  const int tile  = blockIdx.x * 4 + w;
  const int qt    = tile % (S_ / 16);
  const int h     = (tile / (S_ / 16)) % NQ_;
  const int b     = tile / ((S_ / 16) * NQ_);
  const int kvh   = h / (NQ_ / NK_);
  const int qBase = qt * 16;

  // resident Q fragments (16 rows x 128 dims as 4 K-chunks of 32)
  BF16x16 qf[4];
  const bf16* qp = q + (((size_t)b * S_ + qBase + m16) * NQ_ + h) * HD_;
#pragma unroll
  for (int c = 0; c < 4; ++c) {
    qf[c].h[0] = *(const v8bf*)(qp + c * 32 + half * 8);
    qf[c].h[1] = *(const v8bf*)(qp + c * 32 + 16 + half * 8);
  }

  const float NEGINF = -__builtin_inff();
  v8f oacc[8];
  float mi[8], li[8];
#pragma unroll
  for (int e = 0; e < 8; ++e) { oacc[e] = v8f_zero(); mi[e] = NEGINF; li[e] = 0.0f; }

  int j0 = qBase - (WINDOW_ - 1);
  if (j0 < 0) j0 = 0;
  j0 &= ~31;

  for (int jc = j0; jc <= qBase + 15; jc += 32) {
    // prefetch next key chunk's K rows and V rows while this chunk's
    // softmax VALU chain executes (attention is latency-bound per wave)
    const int jn = jc + 32;
    if (jn <= qBase + 15) {
      const bf16* kpf = kk + (((size_t)b * S_ + jn + m16) * NK_ + kvh) * HD_;
      __builtin_prefetch(kpf, 0, 3);
      __builtin_prefetch(kpf + 64, 0, 3);
      const bf16* kpf2 = kpf + (size_t)16 * NK_ * HD_;
      __builtin_prefetch(kpf2, 0, 3);
      __builtin_prefetch(kpf2 + 64, 0, 3);
      const bf16* vpf = vt + (((size_t)b * NK_ + kvh) * HD_ + lane) * S_ + jn;
      __builtin_prefetch(vpf, 0, 3);
      __builtin_prefetch(vpf + (size_t)32 * S_, 0, 3);
      __builtin_prefetch(vpf + (size_t)64 * S_, 0, 3);
      __builtin_prefetch(vpf + (size_t)96 * S_, 0, 3);
    }

    float pv[2][8];
    float rmax[8];
#pragma unroll
    for (int e = 0; e < 8; ++e) rmax[e] = NEGINF;

#pragma unroll
    for (int st = 0; st < 2; ++st) {
      const int kt = jc + st * 16;
      v8f sacc = v8f_zero();
      const bf16* kp = kk + (((size_t)b * S_ + kt + m16) * NK_ + kvh) * HD_;
#pragma unroll
      for (int c = 0; c < 4; ++c) {
        BF16x16 kf;
        kf.h[0] = *(const v8bf*)(kp + c * 32 + half * 8);
        kf.h[1] = *(const v8bf*)(kp + c * 32 + 16 + half * 8);
        sacc = __builtin_amdgcn_wmma_f32_16x16x32_bf16(
            false, qf[c].v, false, kf.v, (short)0, sacc, false, false);
      }
      const int j = kt + m16;
#pragma unroll
      for (int e = 0; e < 8; ++e) {
        const int i = qBase + half * 8 + e;
        float sv = sacc[e] * 0.08838834764831845f;   // 1/sqrt(128)
        sv = 50.0f * tanhf(sv * 0.02f);              // softcap
        sv = ((j <= i) && (j > i - WINDOW_)) ? sv : NEGINF;
        pv[st][e] = sv;
        rmax[e] = fmaxf(rmax[e], sv);
      }
    }

    // row max over the 16 lanes owning this row group (xor<16 keeps the half)
#pragma unroll
    for (int e = 0; e < 8; ++e) {
      float rm = rmax[e];
      rm = fmaxf(rm, __shfl_xor(rm, 1, 32));
      rm = fmaxf(rm, __shfl_xor(rm, 2, 32));
      rm = fmaxf(rm, __shfl_xor(rm, 4, 32));
      rm = fmaxf(rm, __shfl_xor(rm, 8, 32));
      rmax[e] = rm;
    }

    float alpha[8];
#pragma unroll
    for (int e = 0; e < 8; ++e) {
      const float mn = fmaxf(mi[e], rmax[e]);
      alpha[e] = (mi[e] == NEGINF) ? 0.0f : __expf(mi[e] - mn);
      float p0 = (pv[0][e] == NEGINF) ? 0.0f : __expf(pv[0][e] - mn);
      float p1 = (pv[1][e] == NEGINF) ? 0.0f : __expf(pv[1][e] - mn);
      pv[0][e] = p0; pv[1][e] = p1;
      float rs = p0 + p1;
      rs += __shfl_xor(rs, 1, 32);
      rs += __shfl_xor(rs, 2, 32);
      rs += __shfl_xor(rs, 4, 32);
      rs += __shfl_xor(rs, 8, 32);
      li[e] = li[e] * alpha[e] + rs;
      mi[e] = mn;
    }

#pragma unroll
    for (int dt = 0; dt < 8; ++dt)
#pragma unroll
      for (int e = 0; e < 8; ++e) oacc[dt][e] *= alpha[e];

    // C-layout P -> LDS -> A-fragment layout (in-order LDS within one wave)
#pragma unroll
    for (int st = 0; st < 2; ++st)
#pragma unroll
      for (int e = 0; e < 8; ++e)
        ldsP[w][(half * 8 + e) * 32 + st * 16 + m16] = f2bf(pv[st][e]);

    BF16x16 pa;
    pa.h[0] = *(const v8bf*)&ldsP[w][m16 * 32 + half * 8];
    pa.h[1] = *(const v8bf*)&ldsP[w][m16 * 32 + 16 + half * 8];

    // P (16x32) x V (32x128) as 8 N-tiles
#pragma unroll
    for (int dt = 0; dt < 8; ++dt) {
      BF16x16 vf;
      const bf16* vp = vt + (((size_t)b * NK_ + kvh) * HD_ + dt * 16 + m16) * S_ + jc;
      vf.h[0] = *(const v8bf*)(vp + half * 8);
      vf.h[1] = *(const v8bf*)(vp + 16 + half * 8);
      oacc[dt] = __builtin_amdgcn_wmma_f32_16x16x32_bf16(
          false, pa.v, false, vf.v, (short)0, oacc[dt], false, false);
    }
  }

#pragma unroll
  for (int dt = 0; dt < 8; ++dt)
#pragma unroll
    for (int e = 0; e < 8; ++e) {
      const int row = qBase + half * 8 + e;
      o[(((size_t)b * S_ + row) * NQ_ + h) * HD_ + dt * 16 + m16] = oacc[dt][e] / li[e];
    }
}

// ---------------- host launcher ----------------
extern "C" void kernel_launch(void* const* d_in, const int* in_sizes, int n_in,
                              void* d_out, int out_size, void* d_ws, size_t ws_size,
                              hipStream_t stream) {
  (void)in_sizes; (void)n_in; (void)out_size; (void)ws_size;
  const float* x  = (const float*)d_in[0];
  const float* wq = (const float*)d_in[1];
  const float* wk = (const float*)d_in[2];
  const float* wv = (const float*)d_in[3];
  const float* wo = (const float*)d_in[4];
  const float* qn = (const float*)d_in[5];
  const float* kn = (const float*)d_in[6];
  float* out = (float*)d_out;

  const size_t nX  = (size_t)B_ * S_ * HID_;        // 16.7M
  const size_t nWq = (size_t)NQ_ * HD_ * HID_;      // 16.7M
  const size_t nWk = (size_t)NK_ * HD_ * HID_;      // 4.2M
  const size_t nQ  = (size_t)B_ * S_ * NQ_ * HD_;   // 16.7M
  const size_t nK  = (size_t)B_ * S_ * NK_ * HD_;   // 4.2M

  char* ws = (char*)d_ws;
  size_t off = 0;
  auto alloc = [&](size_t bytes) {
    void* p = ws + off;
    off = (off + bytes + 255) & ~(size_t)255;
    return p;
  };
  bf16*  x_bf  = (bf16*)alloc(nX * 2);
  bf16*  wq_bf = (bf16*)alloc(nWq * 2);
  bf16*  wk_bf = (bf16*)alloc(nWk * 2);
  bf16*  wv_bf = (bf16*)alloc(nWk * 2);
  bf16*  wo_bf = (bf16*)alloc(nWq * 2);
  float* q_f   = (float*)alloc(nQ * 4);
  float* k_f   = (float*)alloc(nK * 4);
  float* v_f   = (float*)alloc(nK * 4);
  bf16*  q_bf  = (bf16*)alloc(nQ * 2);
  bf16*  k_bf  = (bf16*)alloc(nK * 2);
  bf16*  v_t   = (bf16*)alloc(nK * 2);
  float* o_f   = (float*)alloc(nQ * 4);
  bf16*  o_bf  = (bf16*)alloc(nQ * 2);

  auto cvt = [&](const float* s, bf16* d, size_t n) {
    f32_to_bf16_kernel<<<(unsigned)((n + 255) / 256), 256, 0, stream>>>(s, d, (int)n);
  };
  cvt(x,  x_bf,  nX);
  cvt(wq, wq_bf, nWq);
  cvt(wk, wk_bf, nWk);
  cvt(wv, wv_bf, nWk);
  cvt(wo, wo_bf, nWq);

  // QKV projections (block tile 128x128)
  gemm_bf16_wmma_kernel<<<dim3(4096 / 128, 4096 / 128), 256, 0, stream>>>(x_bf, wq_bf, q_f, 4096, 4096, 4096);
  gemm_bf16_wmma_kernel<<<dim3(1024 / 128, 4096 / 128), 256, 0, stream>>>(x_bf, wk_bf, k_f, 4096, 1024, 4096);
  gemm_bf16_wmma_kernel<<<dim3(1024 / 128, 4096 / 128), 256, 0, stream>>>(x_bf, wv_bf, v_f, 4096, 1024, 4096);

  // QK-norm + RoPE (-> bf16), V transpose (-> bf16 [B,NK,HD,S])
  qknorm_rope_kernel<<<(B_ * S_ * NQ_) / 4, 128, 0, stream>>>(q_f, q_bf, qn, NQ_);
  qknorm_rope_kernel<<<(B_ * S_ * NK_) / 4, 128, 0, stream>>>(k_f, k_bf, kn, NK_);
  v_transpose_kernel<<<(unsigned)((nK + 255) / 256), 256, 0, stream>>>(v_f, v_t);

  // flash attention
  flash_attn_kernel<<<(B_ * NQ_ * (S_ / 16)) / 4, 128, 0, stream>>>(q_bf, k_bf, v_t, o_f);

  // output projection
  cvt(o_f, o_bf, nQ);
  gemm_bf16_wmma_kernel<<<dim3(4096 / 128, 4096 / 128), 256, 0, stream>>>(o_bf, wo_bf, out, 4096, 4096, 4096);
}
